// GraphModel_88115549045428
// MI455X (gfx1250) — matrix operand order repaired and checked by reading, compiled, not run on hardware
//
#include <hip/hip_runtime.h>

typedef __attribute__((ext_vector_type(16))) _Float16 v16h;
typedef __attribute__((ext_vector_type(8)))  float    v8f;

#define NCOLS 128   // all GEMM outputs are 128 wide (H = OUT = 128)

// ---------------------------------------------------------------------------
// Weight repack: W [K x 128] f32 row-major  ->  f16 WMMA B-fragments.
// B (32x16, 16-bit) lane layout used here (contiguous K, not ISA table order): lane holds N = lane%16,
// K-halves = ks*32 + (lane<16 ? 0 : 16) + h, 16 halves contiguous per lane.
// Fragment index: ((nt*KS + ks)*32 + lane)*16 + h
// ---------------------------------------------------------------------------
__global__ void repack_w_kernel(const float* __restrict__ W,
                                _Float16* __restrict__ Wp, int K) {
    int idx = blockIdx.x * blockDim.x + threadIdx.x;
    int KS = K >> 5;
    int total = 8 * KS * 32;
    if (idx >= total) return;
    int lane = idx & 31;
    int ks   = (idx >> 5) % KS;
    int nt   = (idx >> 5) / KS;
    int col   = nt * 16 + (lane & 15);
    int kbase = ks * 32 + ((lane & 16) ? 16 : 0);
    _Float16* dstp = Wp + (size_t)idx * 16;
#pragma unroll
    for (int h = 0; h < 16; ++h)
        dstp[h] = (_Float16)W[(size_t)(kbase + h) * NCOLS + col];
}

// ---------------------------------------------------------------------------
// c[i] = a[i] + b[i]  (combine the two broadcast biases)
// ---------------------------------------------------------------------------
__global__ void add_vec_kernel(const float* __restrict__ a,
                               const float* __restrict__ b,
                               float* __restrict__ c, int n) {
    int i = blockIdx.x * blockDim.x + threadIdx.x;
    if (i < n) c[i] = a[i] + b[i];
}

// ---------------------------------------------------------------------------
// WMMA GEMM: C[M x 128] = (relu_a ? relu(A) : A)[M x K] @ Wpack + bias
// One wave computes a 16x128 strip: 8 accumulator tiles share one A fragment.
// All 8 B fragments for a K-step are loaded up-front (one clause) so the
// scheduler can overlap the later loads with the earlier WMMAs.
// ---------------------------------------------------------------------------
__global__ __launch_bounds__(128)
void wmma_gemm_kernel(const float* __restrict__ A,
                      const _Float16* __restrict__ Bp,
                      const float* __restrict__ bias,
                      float* __restrict__ C,
                      int M, int K, int relu_a, int has_bias) {
    int wave = threadIdx.x >> 5;
    int lane = threadIdx.x & 31;
    int tm = blockIdx.x * 4 + wave;
    if (tm * 16 >= M) return;            // wave-uniform exit (EXEC stays all-1)

    int KS = K >> 5;
    int row = tm * 16 + (lane & 15);
    if (row >= M) row = M - 1;           // clamp reads for ragged M; stores masked
    int koff = (lane & 16) ? 16 : 0;     // A layout used here (contiguous K): hi half-wave shifts K by 16
    const float* arow = A + (size_t)row * K + koff;
    const v16h* bp = (const v16h*)Bp + lane;

    v8f acc[8];
#pragma unroll
    for (int nt = 0; nt < 8; ++nt) acc[nt] = (v8f){0.f,0.f,0.f,0.f,0.f,0.f,0.f,0.f};

    for (int ks = 0; ks < KS; ++ks) {
        // ---- batch-load all 8 B fragments for this K-step ----
        v16h bf[8];
#pragma unroll
        for (int nt = 0; nt < 8; ++nt) bf[nt] = bp[(nt * KS + ks) * 32];

        // ---- build A fragment (f32 -> f16, optional fused relu) ----
        const float* ap = arow + ks * 32;
        v16h a;
#pragma unroll
        for (int i = 0; i < 8; ++i) {
            float f0 = ap[i];
            float f1 = ap[8 + i];
            if (relu_a) { f0 = fmaxf(f0, 0.f); f1 = fmaxf(f1, 0.f); }
            a[i]     = (_Float16)f0;     // K = base + 0..7
            a[8 + i] = (_Float16)f1;     // K = base + 8..15 (lane-group shifted)
        }

        // ---- 8 WMMAs sharing one A fragment ----
#pragma unroll
        for (int nt = 0; nt < 8; ++nt)
            acc[nt] = __builtin_amdgcn_wmma_f32_16x16x32_f16(
                false, a, false, bf[nt], (short)0, acc[nt], false, false);
    }

    // C/D layout: lane -> col = lane%16, VGPR v -> row = v + (lane<16 ? 0 : 8)
    int col = lane & 15;
    int mrow0 = tm * 16 + ((lane & 16) ? 8 : 0);
    float bb[8];
#pragma unroll
    for (int nt = 0; nt < 8; ++nt)
        bb[nt] = has_bias ? bias[nt * 16 + col] : 0.f;
#pragma unroll
    for (int v = 0; v < 8; ++v) {
        int m = mrow0 + v;
        if (m < M) {
            float* cp = C + (size_t)m * NCOLS + col;
#pragma unroll
            for (int nt = 0; nt < 8; ++nt) cp[nt * 16] = acc[nt][v] + bb[nt];
        }
    }
}

// ---------------------------------------------------------------------------
// Per-node attention scalars: al_s[i] = h_src[i,:] . a_src ; al_d likewise.
// One wave per node, lane handles 4 dims, shfl_xor reduction (wave32).
// ---------------------------------------------------------------------------
__global__ void attn_dots_kernel(const float* __restrict__ hs,
                                 const float* __restrict__ hd,
                                 const float* __restrict__ a_s,
                                 const float* __restrict__ a_d,
                                 float* __restrict__ al_s,
                                 float* __restrict__ al_d, int N) {
    int gid = blockIdx.x * blockDim.x + threadIdx.x;
    int node = gid >> 5, lane = gid & 31;
    if (node >= N) return;
    float4 vs = ((const float4*)a_s)[lane];
    float4 vd = ((const float4*)a_d)[lane];
    float4 h1 = ((const float4*)(hs + (size_t)node * NCOLS))[lane];
    float4 h2 = ((const float4*)(hd + (size_t)node * NCOLS))[lane];
    float ps = h1.x*vs.x + h1.y*vs.y + h1.z*vs.z + h1.w*vs.w;
    float pd = h2.x*vd.x + h2.y*vd.y + h2.z*vd.z + h2.w*vd.w;
#pragma unroll
    for (int off = 16; off > 0; off >>= 1) {
        ps += __shfl_xor(ps, off, 32);
        pd += __shfl_xor(pd, off, 32);
    }
    if (lane == 0) { al_s[node] = ps; al_d[node] = pd; }
}

// ordered-uint encoding so unsigned atomicMax == float max
__device__ __forceinline__ unsigned enc_f(float f) {
    unsigned u = __float_as_uint(f);
    return (u & 0x80000000u) ? ~u : (u | 0x80000000u);
}
__device__ __forceinline__ float dec_f(unsigned u) {
    return (u & 0x80000000u) ? __uint_as_float(u & 0x7FFFFFFFu)
                             : __uint_as_float(~u);
}

__global__ void init_softmax_kernel(unsigned* __restrict__ emax,
                                    float* __restrict__ denom, int N) {
    int i = blockIdx.x * blockDim.x + threadIdx.x;
    if (i < N) { emax[i] = 0u; denom[i] = 0.f; }   // enc(-inf) sentinel
}

__global__ void edge_pass1_kernel(const int* __restrict__ src,
                                  const int* __restrict__ dst,
                                  const float* __restrict__ al_s,
                                  const float* __restrict__ al_d,
                                  float* __restrict__ ebuf,
                                  unsigned* __restrict__ emax, int E) {
    int e = blockIdx.x * blockDim.x + threadIdx.x;
    if (e >= E) return;
    float v = al_s[src[e]] + al_d[dst[e]];
    v = (v > 0.f) ? v : 0.2f * v;        // leaky_relu, slope 0.2
    ebuf[e] = v;
    atomicMax(&emax[dst[e]], enc_f(v));
}

__global__ void edge_pass2_kernel(const int* __restrict__ dst,
                                  const unsigned* __restrict__ emax,
                                  float* __restrict__ ebuf,
                                  float* __restrict__ denom, int E) {
    int e = blockIdx.x * blockDim.x + threadIdx.x;
    if (e >= E) return;
    int d = dst[e];
    float ex = expf(ebuf[e] - dec_f(emax[d]));
    ebuf[e] = ex;                         // in-place: now holds exp term
    atomicAdd(&denom[d], ex);
}

// 32 lanes per edge; each lane scatters 4 dims with f32 atomics
__global__ void edge_aggregate_kernel(const int* __restrict__ src,
                                      const int* __restrict__ dst,
                                      const float* __restrict__ ebuf,
                                      const float* __restrict__ denom,
                                      const float* __restrict__ hsrc,
                                      float* __restrict__ out, int E) {
    int gid = blockIdx.x * blockDim.x + threadIdx.x;
    int e = gid >> 5, lane = gid & 31;
    if (e >= E) return;
    int s = src[e], d = dst[e];
    float alpha = ebuf[e] / denom[d];
    float4 h4 = ((const float4*)(hsrc + (size_t)s * NCOLS))[lane];
    float* op = out + (size_t)d * NCOLS + lane * 4;
    atomicAdd(op + 0, alpha * h4.x);
    atomicAdd(op + 1, alpha * h4.y);
    atomicAdd(op + 2, alpha * h4.z);
    atomicAdd(op + 3, alpha * h4.w);
}

// ---------------------------------------------------------------------------
// Head: per (b,k) pair: mix = relu([x2[pi], proj[pj]] @ W_fc1 + b_fc1);
//       logit = mix . W_fc2 + b_fc2
// ---------------------------------------------------------------------------
__global__ __launch_bounds__(128)
void head_kernel(const float* __restrict__ x2, const float* __restrict__ proj,
                 const int* __restrict__ person_idx,
                 const int* __restrict__ project_idx,
                 const float* __restrict__ Wfc1, const float* __restrict__ bfc1,
                 const float* __restrict__ Wfc2, const float* __restrict__ bfc2,
                 float* __restrict__ out, int Kk) {
    __shared__ float in[2 * NCOLS];
    __shared__ float red[NCOLS];
    int p = blockIdx.x;
    int b = p / Kk;
    int t = threadIdx.x;
    int pi = person_idx[p];
    int pj = project_idx[b];
    in[t]         = x2[(size_t)pi * NCOLS + t];
    in[NCOLS + t] = proj[(size_t)pj * NCOLS + t];
    __syncthreads();
    float acc = bfc1[t];
#pragma unroll 4
    for (int i = 0; i < 2 * NCOLS; ++i) acc += in[i] * Wfc1[i * NCOLS + t];
    acc = fmaxf(acc, 0.f);
    red[t] = acc * Wfc2[t];
    __syncthreads();
#pragma unroll
    for (int s = 64; s > 0; s >>= 1) {
        if (t < s) red[t] += red[t + s];
        __syncthreads();
    }
    if (t == 0) out[p] = red[0] + bfc2[0];
}

// ---------------------------------------------------------------------------
extern "C" void kernel_launch(void* const* d_in, const int* in_sizes, int n_in,
                              void* d_out, int out_size, void* d_ws, size_t ws_size,
                              hipStream_t stream) {
    (void)n_in; (void)out_size; (void)ws_size;

    const float* x_person    = (const float*)d_in[0];
    const int*   edge_index  = (const int*)  d_in[1];
    const float* x_project   = (const float*)d_in[2];
    const int*   person_idx  = (const int*)  d_in[3];
    const int*   project_idx = (const int*)  d_in[4];
    const float* W_src1 = (const float*)d_in[5];
    const float* W_dst1 = (const float*)d_in[6];
    const float* a_src1 = (const float*)d_in[7];
    const float* a_dst1 = (const float*)d_in[8];
    const float* b1     = (const float*)d_in[9];
    const float* Wl1    = (const float*)d_in[10];
    const float* bl1    = (const float*)d_in[11];
    const float* W_src2 = (const float*)d_in[12];
    const float* W_dst2 = (const float*)d_in[13];
    const float* a_src2 = (const float*)d_in[14];
    const float* a_dst2 = (const float*)d_in[15];
    const float* b2     = (const float*)d_in[16];
    const float* Wl2    = (const float*)d_in[17];
    const float* bl2    = (const float*)d_in[18];
    const float* W3     = (const float*)d_in[19];
    const float* b3     = (const float*)d_in[20];
    const float* W_fc1  = (const float*)d_in[21];
    const float* b_fc1  = (const float*)d_in[22];
    const float* W_fc2  = (const float*)d_in[23];
    const float* b_fc2  = (const float*)d_in[24];

    const int D_IN = 256, H = 128;
    int N  = in_sizes[0] / D_IN;
    int E  = in_sizes[1] / 2;
    int NP = in_sizes[2] / 768;
    int BK = in_sizes[3];
    int B  = in_sizes[4];
    int Kk = BK / B;
    const int* src = edge_index;
    const int* dst = edge_index + E;

    // ---- carve workspace (256B aligned) ----
    size_t off = 0;
    auto carve = [&](size_t bytes) -> char* {
        char* p = (char*)d_ws + off;
        off += (bytes + 255) & ~(size_t)255;
        return p;
    };
    float*    hsrc   = (float*)   carve((size_t)N * H * 4);
    float*    hdst   = (float*)   carve((size_t)N * H * 4);
    float*    x1     = (float*)   carve((size_t)N * H * 4);
    float*    x2     = (float*)   carve((size_t)N * H * 4);
    float*    proj   = (float*)   carve((size_t)NP * H * 4);
    float*    als    = (float*)   carve((size_t)N * 4);
    float*    ald    = (float*)   carve((size_t)N * 4);
    unsigned* emax   = (unsigned*)carve((size_t)N * 4);
    float*    denom  = (float*)   carve((size_t)N * 4);
    float*    ebuf   = (float*)   carve((size_t)E * 4);
    _Float16* wp_s1  = (_Float16*)carve((size_t)D_IN * H * 2);
    _Float16* wp_d1  = (_Float16*)carve((size_t)D_IN * H * 2);
    _Float16* wp_l1  = (_Float16*)carve((size_t)D_IN * H * 2);
    _Float16* wp_s2  = (_Float16*)carve((size_t)H * H * 2);
    _Float16* wp_d2  = (_Float16*)carve((size_t)H * H * 2);
    _Float16* wp_l2  = (_Float16*)carve((size_t)H * H * 2);
    _Float16* wp_3   = (_Float16*)carve((size_t)768 * H * 2);
    float*    bias1  = (float*)   carve(H * 4);
    float*    bias2  = (float*)   carve(H * 4);

    // ---- weight repacks + bias combines ----
    auto repack = [&](const float* W, _Float16* Wp, int K) {
        int th = (K >> 5) * 8 * 32;
        repack_w_kernel<<<(th + 255) / 256, 256, 0, stream>>>(W, Wp, K);
    };
    repack(W_src1, wp_s1, D_IN);
    repack(W_dst1, wp_d1, D_IN);
    repack(Wl1,    wp_l1, D_IN);
    repack(W_src2, wp_s2, H);
    repack(W_dst2, wp_d2, H);
    repack(Wl2,    wp_l2, H);
    repack(W3,     wp_3,  768);
    add_vec_kernel<<<1, H, 0, stream>>>(b1, bl1, bias1, H);
    add_vec_kernel<<<1, H, 0, stream>>>(b2, bl2, bias2, H);

    auto gemm = [&](const float* A, const _Float16* Bp, const float* bias,
                    float* C, int M, int K, int relu_a, int has_bias) {
        int tiles = (M + 15) / 16;
        int blocks = (tiles + 3) / 4;
        wmma_gemm_kernel<<<blocks, 128, 0, stream>>>(A, Bp, bias, C, M, K,
                                                     relu_a, has_bias);
    };

    int eb  = (E + 255) / 256;                 // edge-parallel blocks
    int eb32 = ((size_t)E * 32 + 255) / 256;   // 32-lanes-per-edge blocks
    int nb32 = ((size_t)N * 32 + 255) / 256;   // wave-per-node blocks

    // ======== GAT layer 1 ========
    gemm(x_person, wp_s1, b1,    hsrc, N, D_IN, 0, 0);
    gemm(x_person, wp_d1, b1,    hdst, N, D_IN, 0, 0);
    gemm(x_person, wp_l1, bias1, x1,   N, D_IN, 0, 1);   // residual + (bl1+b1)
    attn_dots_kernel<<<nb32, 256, 0, stream>>>(hsrc, hdst, a_src1, a_dst1,
                                               als, ald, N);
    init_softmax_kernel<<<(N + 255) / 256, 256, 0, stream>>>(emax, denom, N);
    edge_pass1_kernel<<<eb, 256, 0, stream>>>(src, dst, als, ald, ebuf, emax, E);
    edge_pass2_kernel<<<eb, 256, 0, stream>>>(dst, emax, ebuf, denom, E);
    edge_aggregate_kernel<<<eb32, 256, 0, stream>>>(src, dst, ebuf, denom,
                                                    hsrc, x1, E);

    // ======== GAT layer 2 (relu fused into A loads) ========
    gemm(x1, wp_s2, b2,    hsrc, N, H, 1, 0);
    gemm(x1, wp_d2, b2,    hdst, N, H, 1, 0);
    gemm(x1, wp_l2, bias2, x2,   N, H, 1, 1);            // residual + (bl2+b2)
    attn_dots_kernel<<<nb32, 256, 0, stream>>>(hsrc, hdst, a_src2, a_dst2,
                                               als, ald, N);
    init_softmax_kernel<<<(N + 255) / 256, 256, 0, stream>>>(emax, denom, N);
    edge_pass1_kernel<<<eb, 256, 0, stream>>>(src, dst, als, ald, ebuf, emax, E);
    edge_pass2_kernel<<<eb, 256, 0, stream>>>(dst, emax, ebuf, denom, E);
    edge_aggregate_kernel<<<eb32, 256, 0, stream>>>(src, dst, ebuf, denom,
                                                    hsrc, x2, E);

    // ======== project branch + head ========
    gemm(x_project, wp_3, b3, proj, NP, 768, 0, 1);
    head_kernel<<<BK, 128, 0, stream>>>(x2, proj, person_idx, project_idx,
                                        W_fc1, b_fc1, W_fc2, b_fc2,
                                        (float*)d_out, Kk);
}
